// MSMTTEGRUAttModel_83176336654769
// MI455X (gfx1250) — matrix operand order repaired and checked by reading, not compiled, not run on hardware
//
#include <hip/hip_runtime.h>
#include <hip/hip_bf16.h>
#include <cstdint>

typedef __attribute__((ext_vector_type(16))) _Float16 v16h;
typedef __attribute__((ext_vector_type(8)))  float    v8f;

#define Bsz  1024
#define Tsz  256
#define Hsz  128
#define H3   384

union HPack { uint32_t u; _Float16 h[2]; };
union AFrag { v16h v; uint32_t u[8]; };
union H16   { _Float16 h; uint16_t u; };

// ---- fragment loaders (layouts per CDNA5 ISA §7.12.2) ----------

// A (16x32 f16), source: LDS row-major halves [16][128], K-tile base kb
__device__ inline v16h load_a_frag_lds(const uint16_t* h16, int lane, int kb) {
  const int m   = lane & 15;
  const int hi8 = (lane >> 4) * 8;
  AFrag f;
#pragma unroll
  for (int v = 0; v < 8; ++v) {
    const int k = kb + ((v & 4) ? 16 : 0) + hi8 + 2 * (v & 3);
    f.u[v] = *(const uint32_t*)(&h16[m * Hsz + k]);
  }
  return f.v;
}

// A (16x32 f16), source: global f32 row-major [M][ldk]
__device__ inline v16h load_a_frag_f32(const float* A, int ldk, int row0, int kb, int lane) {
  const int m   = row0 + (lane & 15);
  const int hi8 = (lane >> 4) * 8;
  AFrag f;
#pragma unroll
  for (int v = 0; v < 8; ++v) {
    const int k = kb + ((v & 4) ? 16 : 0) + hi8 + 2 * (v & 3);
    HPack p;
    p.h[0] = (_Float16)A[(size_t)m * ldk + k];
    p.h[1] = (_Float16)A[(size_t)m * ldk + k + 1];
    f.u[v] = p.u;
  }
  return f.v;
}

// B (32x16 f16), source: global f32 row-major [K][ldn]; lane N-striped, K in table order (same as A)
__device__ inline v16h load_b_frag_f32(const float* W, int ldn, int kb, int nb, int lane) {
  const int n    = nb + (lane & 15);
  const int kadd = (lane >> 4) * 8;
  AFrag f;
#pragma unroll
  for (int v = 0; v < 8; ++v) {
    const int k = kb + ((v & 4) ? 16 : 0) + kadd + 2 * (v & 3);
    HPack p;
    p.h[0] = (_Float16)W[(size_t)k * ldn + n];
    p.h[1] = (_Float16)W[(size_t)(k + 1) * ldn + n];
    f.u[v] = p.u;
  }
  return f.v;
}

__device__ inline float sigm(float x) { return 1.f / (1.f + __expf(-x)); }
__device__ inline float tanh_fast(float x) {
  float e = __expf(-2.f * x);
  return (1.f - e) / (1.f + e);
}

// ---- kernel 1: bake input-projection lookup tables ------------------------
// The embedding gather commutes with the input projection:
//   E_time[hour_idx] @ Wh_x + bh[0]  ==  (E_time @ Wh_x + bh[0])[hour_idx]
// so both big [B*T,128]@[128,384] GEMMs collapse to 24x384 / 7x384 tables.
__global__ void xp_tables_kernel(const float* __restrict__ E_time, const float* __restrict__ E_week,
                                 const float* __restrict__ Wh_x, const float* __restrict__ bh,
                                 const float* __restrict__ Ww_x, const float* __restrict__ bw,
                                 float* __restrict__ XPt, float* __restrict__ XPw) {
  int i = blockIdx.x * blockDim.x + threadIdx.x;
  if (i < 24 * H3) {
    int r = i / H3, n = i % H3;
    float acc = bh[n];
#pragma unroll 8
    for (int k = 0; k < Hsz; ++k) acc += E_time[r * Hsz + k] * Wh_x[k * H3 + n];
    XPt[i] = acc;
  } else if (i < (24 + 7) * H3) {
    int i2 = i - 24 * H3;
    int r = i2 / H3, n = i2 % H3;
    float acc = bw[n];
#pragma unroll 8
    for (int k = 0; k < Hsz; ++k) acc += E_week[r * Hsz + k] * Ww_x[k * H3 + n];
    XPw[i2] = acc;
  }
}

// ---- kernel 2: persistent batch-tile GRU ----------------------------------
// grid = (64 batch tiles of 16 rows, 3 GRUs); block = 256 threads = 8 waves.
// Wave w owns hp columns [48w, 48w+48): 3 N-tiles, K=128 => 12 WMMAs/step.
// Recurrent weight B-fragments live in registers for all 256 steps; all
// per-step inputs (idx / spatial) are hoisted to LDS before the T loop so the
// steady-state loop reads global memory only for XPt/XPw rows.
union Stage { uint8_t idx8[16 * Tsz]; _Float16 sp16[16 * Tsz * 2]; };

__global__ __launch_bounds__(256) void gru_kernel(
    const float* __restrict__ spatial, const int* __restrict__ hour_idx,
    const int* __restrict__ week_idx,
    const float* __restrict__ XPt, const float* __restrict__ XPw,
    const float* __restrict__ Ws_x,
    const float* __restrict__ bs, const float* __restrict__ bh, const float* __restrict__ bw,
    const float* __restrict__ Ws_h, const float* __restrict__ Wh_h, const float* __restrict__ Ww_h,
    float* __restrict__ F) {
  const int gru  = blockIdx.y;
  const int row0 = blockIdx.x * 16;
  const int tid  = threadIdx.x;
  const int lane = tid & 31;
  const int wave = tid >> 5;

  const float* Whh = (gru == 0) ? Ws_h : (gru == 1 ? Wh_h : Ww_h);
  const float* b1  = ((gru == 0) ? bs : (gru == 1 ? bh : bw)) + H3;

  __shared__ uint16_t h16[16 * Hsz];   // f16 mirror of h (A operand source)
  __shared__ float    hf [16 * Hsz];   // f32 h state
  __shared__ float    hp [16 * H3];    // h @ Wh per step
  __shared__ float    b1s[H3];
  __shared__ float    wsx_s[2 * H3];
  __shared__ float    bs0_s[H3];
  __shared__ Stage    stage;           // 16KB: all per-step inputs, preloaded

  for (int i = tid; i < H3; i += 256) b1s[i] = b1[i];
  if (gru == 0) {
    for (int i = tid; i < 2 * H3; i += 256) wsx_s[i] = Ws_x[i];
    for (int i = tid; i < H3; i += 256) bs0_s[i] = bs[i];
    // spatial inputs for this tile, f16-packed: [m][t][2]
    for (int i = tid; i < 16 * Tsz; i += 256) {
      const int m = i >> 8, t = i & (Tsz - 1);
      const size_t g = ((size_t)(row0 + m) * Tsz + t) * 2;
      stage.sp16[i * 2]     = (_Float16)spatial[g];
      stage.sp16[i * 2 + 1] = (_Float16)spatial[g + 1];
    }
  } else {
    const int* idx = (gru == 1) ? hour_idx : week_idx;
    for (int i = tid; i < 16 * Tsz; i += 256) {
      const int m = i >> 8, t = i & (Tsz - 1);
      stage.idx8[i] = (uint8_t)idx[(size_t)(row0 + m) * Tsz + t];
    }
  }
  for (int i = tid; i < 16 * Hsz; i += 256) { hf[i] = 0.f; h16[i] = 0; }

  // Preload recurrent-weight fragments: 3 N-tiles x 4 K-tiles, live all T steps.
  const int nb0 = wave * 48;
  v16h bfr[3][4];
#pragma unroll
  for (int nt = 0; nt < 3; ++nt)
#pragma unroll
    for (int kt = 0; kt < 4; ++kt)
      bfr[nt][kt] = load_b_frag_f32(Whh, H3, kt * 32, nb0 + nt * 16, lane);

  __syncthreads();

  const int m_g = tid >> 4;          // gate-phase row
  const int j0  = (tid & 15) * 8;    // gate-phase col base
  const int nc  = lane & 15;
  const int hi8 = (lane >> 4) * 8;

  for (int t = 0; t < Tsz; ++t) {
    // hp = h @ Whh  (f16 WMMA, f32 accumulate)
    v8f c0 = {}, c1 = {}, c2 = {};
#pragma unroll
    for (int kt = 0; kt < 4; ++kt) {
      v16h a = load_a_frag_lds(h16, lane, kt * 32);
      c0 = __builtin_amdgcn_wmma_f32_16x16x32_f16(false, a, false, bfr[0][kt], (short)0, c0, false, false);
      c1 = __builtin_amdgcn_wmma_f32_16x16x32_f16(false, a, false, bfr[1][kt], (short)0, c1, false, false);
      c2 = __builtin_amdgcn_wmma_f32_16x16x32_f16(false, a, false, bfr[2][kt], (short)0, c2, false, false);
    }
#pragma unroll
    for (int r = 0; r < 8; ++r) {
      hp[(r + hi8) * H3 + nb0 +      nc] = c0[r];
      hp[(r + hi8) * H3 + nb0 + 16 + nc] = c1[r];
      hp[(r + hi8) * H3 + nb0 + 32 + nc] = c2[r];
    }
    __syncthreads();

    // gate update (Keras reset_after=True, gate order z,r,h)
    float s0 = 0.f, s1 = 0.f;
    const float* xprow = nullptr;
    if (gru == 0) {
      HPack p;
      p.u = *(const uint32_t*)(&stage.sp16[(m_g * Tsz + t) * 2]);
      s0 = (float)p.h[0];
      s1 = (float)p.h[1];
    } else {
      const int id = (int)stage.idx8[m_g * Tsz + t];
      xprow = ((gru == 1) ? XPt : XPw) + id * H3;
    }

#pragma unroll
    for (int jj = 0; jj < 8; ++jj) {
      const int j = j0 + jj;
      float xz, xr, xh;
      if (gru == 0) {
        xz = s0 * wsx_s[j]        + s1 * wsx_s[H3 + j]        + bs0_s[j];
        xr = s0 * wsx_s[128 + j]  + s1 * wsx_s[H3 + 128 + j]  + bs0_s[128 + j];
        xh = s0 * wsx_s[256 + j]  + s1 * wsx_s[H3 + 256 + j]  + bs0_s[256 + j];
      } else {
        xz = xprow[j]; xr = xprow[128 + j]; xh = xprow[256 + j];
      }
      const float hz = hp[m_g * H3 + j]        + b1s[j];
      const float hr = hp[m_g * H3 + 128 + j]  + b1s[128 + j];
      const float hn = hp[m_g * H3 + 256 + j]  + b1s[256 + j];
      const float z  = sigm(xz + hz);
      const float r  = sigm(xr + hr);
      const float hh = tanh_fast(xh + r * hn);
      const float ho = hf[m_g * Hsz + j];
      const float hnew = z * ho + (1.f - z) * hh;
      hf[m_g * Hsz + j] = hnew;
      H16 cv; cv.h = (_Float16)hnew;
      h16[m_g * Hsz + j] = cv.u;
    }
    __syncthreads();
  }

  for (int i = tid; i < 16 * Hsz; i += 256)
    F[(size_t)gru * Bsz * Hsz + (size_t)(row0 + i / Hsz) * Hsz + (i & (Hsz - 1))] = hf[i];
}

// ---- kernel 3: attention fusion over the 3 feature channels ---------------
__global__ void attn_kernel(const float* __restrict__ F, const float* __restrict__ Wa,
                            const float* __restrict__ ba, float* __restrict__ sf) {
  int i = blockIdx.x * blockDim.x + threadIdx.x;
  if (i >= Bsz * Hsz) return;
  const float f0 = F[i];
  const float f1 = F[Bsz * Hsz + i];
  const float f2 = F[2 * Bsz * Hsz + i];
  float s[3];
#pragma unroll
  for (int c = 0; c < 3; ++c)
    s[c] = fmaxf(f0 * Wa[c] + f1 * Wa[3 + c] + f2 * Wa[6 + c] + ba[c], 0.f);
  const float mx = fmaxf(s[0], fmaxf(s[1], s[2]));
  const float e0 = __expf(s[0] - mx), e1 = __expf(s[1] - mx), e2 = __expf(s[2] - mx);
  const float inv = 1.f / (e0 + e1 + e2);
  sf[i] = (f0 * e0 + f1 * e1 + f2 * e2) * inv;
}

// ---- kernel 4: WMMA GEMM layer: C = relu(A[M,K] @ W[K,N] + bias) ----------
// block = 8 waves; wave computes a 16x64 tile. grid = (N/64, M/128).
__global__ __launch_bounds__(256) void gemm_wmma(const float* __restrict__ A,
                                                 const float* __restrict__ W,
                                                 const float* __restrict__ bias,
                                                 float* __restrict__ C,
                                                 int M, int K, int N, int relu) {
  const int lane  = threadIdx.x & 31;
  const int wave  = threadIdx.x >> 5;
  const int row0  = blockIdx.y * 128 + wave * 16;
  const int nbase = blockIdx.x * 64;

  v8f c[4] = {v8f{}, v8f{}, v8f{}, v8f{}};
  for (int kb = 0; kb < K; kb += 32) {
    const v16h a = load_a_frag_f32(A, K, row0, kb, lane);
#pragma unroll
    for (int nt = 0; nt < 4; ++nt) {
      const v16h b = load_b_frag_f32(W, N, kb, nbase + nt * 16, lane);
      c[nt] = __builtin_amdgcn_wmma_f32_16x16x32_f16(false, a, false, b, (short)0, c[nt], false, false);
    }
  }

  const int nc  = lane & 15;
  const int hi8 = (lane >> 4) * 8;
#pragma unroll
  for (int nt = 0; nt < 4; ++nt) {
    const int n  = nbase + nt * 16 + nc;
    const float bv = bias[n];
#pragma unroll
    for (int r = 0; r < 8; ++r) {
      float v = c[nt][r] + bv;
      if (relu) v = fmaxf(v, 0.f);
      C[(size_t)(row0 + r + hi8) * N + n] = v;
    }
  }
}

// ---- kernel 5: residual + Dense(1): out[b] = (x4[b]+sf[b]) . Wo + bo ------
__global__ void out_kernel(const float* __restrict__ x4, const float* __restrict__ sf,
                           const float* __restrict__ Wo, const float* __restrict__ bo,
                           float* __restrict__ out) {
  const int lane = threadIdx.x & 31;
  const int wave = threadIdx.x >> 5;
  const int row  = blockIdx.x * 8 + wave;
  float acc = 0.f;
#pragma unroll
  for (int j = lane; j < Hsz; j += 32)
    acc += (x4[(size_t)row * Hsz + j] + sf[(size_t)row * Hsz + j]) * Wo[j];
#pragma unroll
  for (int off = 16; off > 0; off >>= 1) acc += __shfl_down(acc, off, 32);
  if (lane == 0) out[row] = acc + bo[0];
}

// ---------------------------------------------------------------------------
extern "C" void kernel_launch(void* const* d_in, const int* in_sizes, int n_in,
                              void* d_out, int out_size, void* d_ws, size_t ws_size,
                              hipStream_t stream) {
  const float* spatial  = (const float*)d_in[0];
  const int*   hour_idx = (const int*)  d_in[1];
  const int*   week_idx = (const int*)  d_in[2];
  const float* E_time   = (const float*)d_in[3];
  const float* E_week   = (const float*)d_in[4];
  const float* Ws_x = (const float*)d_in[5];
  const float* Ws_h = (const float*)d_in[6];
  const float* bs   = (const float*)d_in[7];
  const float* Wh_x = (const float*)d_in[8];
  const float* Wh_h = (const float*)d_in[9];
  const float* bh   = (const float*)d_in[10];
  const float* Ww_x = (const float*)d_in[11];
  const float* Ww_h = (const float*)d_in[12];
  const float* bw   = (const float*)d_in[13];
  const float* Wa   = (const float*)d_in[14];
  const float* ba   = (const float*)d_in[15];
  const float* W1   = (const float*)d_in[16];
  const float* b1   = (const float*)d_in[17];
  const float* W2   = (const float*)d_in[18];
  const float* b2   = (const float*)d_in[19];
  const float* W3   = (const float*)d_in[20];
  const float* b3   = (const float*)d_in[21];
  const float* W4   = (const float*)d_in[22];
  const float* b4   = (const float*)d_in[23];
  const float* Wo   = (const float*)d_in[24];
  const float* bo   = (const float*)d_in[25];
  float* out = (float*)d_out;

  float* ws  = (float*)d_ws;
  float* XPt = ws; ws += 24 * H3;
  float* XPw = ws; ws += 7 * H3;
  float* F   = ws; ws += 3 * Bsz * Hsz;
  float* sf  = ws; ws += Bsz * Hsz;
  float* x1  = ws; ws += Bsz * 1024;
  float* x2  = ws; ws += Bsz * 512;
  float* x3  = ws; ws += Bsz * 256;
  float* x4  = ws; ws += Bsz * Hsz;

  // 1) bake input-projection lookup tables (gather commutes with matmul)
  xp_tables_kernel<<<47, 256, 0, stream>>>(E_time, E_week, Wh_x, bh, Ww_x, bw, XPt, XPw);

  // 2) persistent batch-tile GRUs (3 GRUs x 64 tiles of 16 rows)
  gru_kernel<<<dim3(64, 3), 256, 0, stream>>>(spatial, hour_idx, week_idx, XPt, XPw,
                                              Ws_x, bs, bh, bw, Ws_h, Wh_h, Ww_h, F);

  // 3) attention fusion
  attn_kernel<<<(Bsz * Hsz) / 256, 256, 0, stream>>>(F, Wa, ba, sf);

  // 4) MLP (WMMA GEMMs)
  gemm_wmma<<<dim3(1024 / 64, Bsz / 128), 256, 0, stream>>>(sf, W1, b1, x1, Bsz, 128, 1024, 1);
  gemm_wmma<<<dim3(512 / 64,  Bsz / 128), 256, 0, stream>>>(x1, W2, b2, x2, Bsz, 1024, 512, 1);
  gemm_wmma<<<dim3(256 / 64,  Bsz / 128), 256, 0, stream>>>(x2, W3, b3, x3, Bsz, 512, 256, 1);
  gemm_wmma<<<dim3(128 / 64,  Bsz / 128), 256, 0, stream>>>(x3, W4, b4, x4, Bsz, 256, 128, 1);

  // 5) residual + Dense(1)
  out_kernel<<<Bsz / 8, 256, 0, stream>>>(x4, sf, Wo, bo, out);
}
